// MultiHeadAttention_64871186039330
// MI455X (gfx1250) — matrix operand order repaired and checked by reading, not compiled, not run on hardware
//
#include <hip/hip_runtime.h>

typedef __attribute__((ext_vector_type(16))) __bf16        v16bf;
typedef __attribute__((ext_vector_type(8)))  float         v8f;
typedef __attribute__((ext_vector_type(4)))  unsigned int  v4u;

#define BATCH 2
#define SEQ   2048
#define EMB   1024
#define HEADS 16
#define HD    64
#define QKV3  3072   // 3*EMB

// ---------- helpers ----------

static __device__ __forceinline__ unsigned short f2bf(float f) {
  unsigned u = __float_as_uint(f);
  u += 0x7FFFu + ((u >> 16) & 1u);          // round to nearest even
  return (unsigned short)(u >> 16);
}

// 8 bf16 from p0 -> VGPR0-3 half, 8 bf16 from p1 -> VGPR4-7 half
static __device__ __forceinline__ v16bf load_frag(const unsigned short* p0,
                                                  const unsigned short* p1) {
  union { v4u u[2]; v16bf v; } f;
  f.u[0] = *(const v4u*)p0;
  f.u[1] = *(const v4u*)p1;
  return f.v;
}

static __device__ __forceinline__ v16bf ones_frag() {
  union { v4u u[2]; v16bf v; } f;
  f.u[0] = v4u{0x3F803F80u, 0x3F803F80u, 0x3F803F80u, 0x3F803F80u};
  f.u[1] = f.u[0];
  return f.v;
}

static __device__ __forceinline__ v8f wmma_bf16(v16bf a, v16bf b, v8f c) {
  return __builtin_amdgcn_wmma_f32_16x16x32_bf16(false, a, false, b,
                                                 (short)0, c, false, false);
}

// LDS generic pointers carry the LDS byte offset in addr[31:0] (ISA 10.2 aperture map)
static __device__ __forceinline__ unsigned lds_off(const void* p) {
  return (unsigned)(unsigned long long)p;
}

// per-lane 16-byte async copy global -> LDS (ASYNCcnt-tracked)
static __device__ __forceinline__ void async_g2l_b128(unsigned lds_byte_off,
                                                      const void* gptr) {
  asm volatile("global_load_async_to_lds_b128 %0, %1, off"
               :: "v"(lds_byte_off), "v"((unsigned long long)gptr)
               : "memory");
}
static __device__ __forceinline__ void wait_async0() {
  asm volatile("s_wait_asynccnt 0x0" ::: "memory");
}

// max-reduction over the 16 lanes of a half-wave (C/D frag rows live per half)
static __device__ __forceinline__ float hredmax(float v) {
  v = fmaxf(v, __shfl_xor(v, 1, 32));
  v = fmaxf(v, __shfl_xor(v, 2, 32));
  v = fmaxf(v, __shfl_xor(v, 4, 32));
  v = fmaxf(v, __shfl_xor(v, 8, 32));
  return v;
}

// ---------- fp32 -> bf16 pack (2 elements / thread) ----------

__global__ void __launch_bounds__(256)
f32_to_bf16_kernel(const float* __restrict__ src, unsigned int* __restrict__ dst, int npair) {
  int i = blockIdx.x * blockDim.x + threadIdx.x;
  if (i < npair) {
    float2 v = ((const float2*)src)[i];
    dst[i] = (unsigned)f2bf(v.x) | ((unsigned)f2bf(v.y) << 16);
  }
}

// ---------- WMMA GEMM: C[M,N] = A[M,K](bf16,row) * W[N,K](bf16,row)^T + bias ----------
// block: 256 threads (8 waves); block tile 128x64; wave tile 16x64; K-step 64.
// W tile staged in LDS via async loads, shared by all 8 waves.

#define WLS 72   // padded LDS row stride (elements): 36-bank stride, conflict-free b128

template <bool OUT_BF16>
__global__ void __launch_bounds__(256)
gemm_wmma_kernel(const unsigned short* __restrict__ A,
                 const unsigned short* __restrict__ W,
                 const float* __restrict__ bias,
                 void* __restrict__ Cout,
                 int M, int N, int K) {
  __shared__ __align__(16) unsigned short lds_w[64 * WLS];

  const int lane = threadIdx.x & 31;
  const int wave = threadIdx.x >> 5;       // 0..7
  const int ln = lane & 15;
  const int hi = lane >> 4;
  const int m0 = blockIdx.y * 128 + wave * 16;
  const int n0 = blockIdx.x * 64;

  v8f acc[4];
#pragma unroll
  for (int t = 0; t < 4; ++t) acc[t] = v8f{0.f,0.f,0.f,0.f,0.f,0.f,0.f,0.f};

  const unsigned short* ap = A + (size_t)(m0 + ln) * K + hi * 16;

  // async staging assignment: 64 rows x 64 k-elements per step, 2 x 16B per thread
  const int srow  = threadIdx.x >> 2;        // 0..63
  const int spart = (threadIdx.x & 3) * 2;   // 16B chunk index {0,2,4,6}
  const unsigned short* wsrc = W + (size_t)(n0 + srow) * K + spart * 8;
  const unsigned wdst = lds_off(&lds_w[srow * WLS]) + spart * 16;

  for (int k0 = 0; k0 < K; k0 += 64) {
    async_g2l_b128(wdst,      wsrc + k0);
    async_g2l_b128(wdst + 16, wsrc + k0 + 8);

    v16bf a0 = load_frag(ap + k0,      ap + k0 + 8);
    v16bf a1 = load_frag(ap + k0 + 32, ap + k0 + 40);
    __builtin_prefetch(ap + k0 + 512, 0, 0);   // global_prefetch_b8

    wait_async0();
    __syncthreads();
#pragma unroll
    for (int t = 0; t < 4; ++t) {
      const unsigned short* wr = lds_w + (t * 16 + ln) * WLS + hi * 16;
      acc[t] = wmma_bf16(a0, load_frag(wr, wr + 8), acc[t]);
      acc[t] = wmma_bf16(a1, load_frag(wr + 32, wr + 40), acc[t]);
    }
    __syncthreads();   // protect lds_w before next async stage
  }

#pragma unroll
  for (int t = 0; t < 4; ++t) {
    const int n = n0 + t * 16 + ln;
    const float bn = bias[n];
#pragma unroll
    for (int r = 0; r < 8; ++r) {
      const int m = m0 + r + hi * 8;
      const float val = acc[t][r] + bn;
      if (OUT_BF16)
        ((unsigned short*)Cout)[(size_t)m * N + n] = f2bf(val);
      else
        ((float*)Cout)[(size_t)m * N + n] = val;
    }
  }
}

// ---------- Flash attention over the reshaped qkv ----------
// Q/K/V element (b,h,r,c) = qkv[(b*2048 + h*128 + (r>>4))*3072 + (r&15)*192 + sel*64 + c]
// grid: (qtile 0..31, h 0..15, b 0..1); block 128 (4 waves); wave owns 16 query rows.

#define VTS 40   // padded LDS row stride (elements): 20-bank stride, conflict-free b128

__global__ void __launch_bounds__(128)
attn_flash_kernel(const unsigned short* __restrict__ qkv,
                  unsigned short* __restrict__ ctx) {
  __shared__ __align__(16) unsigned short lds_p[4][16 * VTS];   // P re-layout, per wave
  __shared__ __align__(16) unsigned short lds_vt[HD * VTS];     // V tile, transposed
  __shared__ __align__(16) unsigned short lds_k[32 * WLS];      // K tile, row-major (async)

  const int lane = threadIdx.x & 31;
  const int wave = threadIdx.x >> 5;
  const int ln = lane & 15;
  const int hi = lane >> 4;
  const int qt = blockIdx.x;   // 64-row query block
  const int h  = blockIdx.y;
  const int b  = blockIdx.z;

  const unsigned short* base = qkv + (size_t)(b * SEQ + h * 128) * QKV3;
  const int m0 = qt * 64 + wave * 16;

  // Q A-frags (c in [0,32) and [32,64)) — all 16 rows of this tile share (r>>4)
  const unsigned short* qrow = base + (size_t)(m0 >> 4) * QKV3 + ln * 192;
  const v16bf aq0 = load_frag(qrow +      hi * 16, qrow + 8 + hi * 16);
  const v16bf aq1 = load_frag(qrow + 32 + hi * 16, qrow + 40 + hi * 16);
  const v16bf bones = ones_frag();

  float rmax[8], rsum[8];
  v8f acc[4];
#pragma unroll
  for (int r = 0; r < 8; ++r) { rmax[r] = -1e30f; rsum[r] = 0.f; }
#pragma unroll
  for (int t = 0; t < 4; ++t) acc[t] = v8f{0.f,0.f,0.f,0.f,0.f,0.f,0.f,0.f};

  // staging assignments (32 kv rows per tile)
  const int srow  = threadIdx.x >> 2;        // 0..31
  const int spart = (threadIdx.x & 3) * 2;   // 16B chunk index {0,2,4,6}

  for (int kv0 = 0; kv0 < SEQ; kv0 += 32) {
    const int rk = kv0 + srow;
    const unsigned short* krow =
        base + (size_t)(rk >> 4) * QKV3 + (rk & 15) * 192 + 64 + spart * 8;
    const unsigned kdst = lds_off(&lds_k[srow * WLS]) + spart * 16;
    async_g2l_b128(kdst,      krow);
    async_g2l_b128(kdst + 16, krow + 8);

    // ---- cooperative V tile stage, transposed: lds_vt[c][kv_local] ----
    {
      const int j  = threadIdx.x >> 2;        // kv row 0..31
      const int c0 = (threadIdx.x & 3) * 16;  // head-dim start
      const unsigned short* vp =
          base + (size_t)(rk >> 4) * QKV3 + (rk & 15) * 192 + 128 + c0;
      v4u d0 = *(const v4u*)vp;
      v4u d1 = *(const v4u*)(vp + 8);
#pragma unroll
      for (int q = 0; q < 4; ++q) {
        lds_vt[(c0 + 2 * q    ) * VTS + j] = (unsigned short)(d0[q] & 0xffffu);
        lds_vt[(c0 + 2 * q + 1) * VTS + j] = (unsigned short)(d0[q] >> 16);
        lds_vt[(c0 + 8 + 2 * q    ) * VTS + j] = (unsigned short)(d1[q] & 0xffffu);
        lds_vt[(c0 + 8 + 2 * q + 1) * VTS + j] = (unsigned short)(d1[q] >> 16);
      }
    }
    wait_async0();
    __syncthreads();

    // ---- scores: S[16 x 32] = Q * K^T  (K B-frags from async-staged LDS tile) ----
    const v8f z = v8f{0.f,0.f,0.f,0.f,0.f,0.f,0.f,0.f};
    const unsigned short* k0p = lds_k + ln * WLS + hi * 16;          // rows 0..15
    const unsigned short* k1p = lds_k + (16 + ln) * WLS + hi * 16;   // rows 16..31
    v8f s0 = wmma_bf16(aq0, load_frag(k0p, k0p + 8), z);
    s0     = wmma_bf16(aq1, load_frag(k0p + 32, k0p + 40), s0);
    v8f s1 = wmma_bf16(aq0, load_frag(k1p, k1p + 8), z);
    s1     = wmma_bf16(aq1, load_frag(k1p + 32, k1p + 40), s1);

    // ---- online softmax: shuffle tree only for the row max ----
    float mnew[8], corr[8];
#pragma unroll
    for (int r = 0; r < 8; ++r) {
      s0[r] *= 0.125f;                       // 1/sqrt(64)
      s1[r] *= 0.125f;
      float t = hredmax(fmaxf(s0[r], s1[r]));
      mnew[r] = fmaxf(rmax[r], t);
      corr[r] = __expf(rmax[r] - mnew[r]);
      rmax[r] = mnew[r];
    }
#pragma unroll
    for (int r = 0; r < 8; ++r) {
      s0[r] = __expf(s0[r] - mnew[r]);
      s1[r] = __expf(s1[r] - mnew[r]);
    }
#pragma unroll
    for (int t = 0; t < 4; ++t)
#pragma unroll
      for (int r = 0; r < 8; ++r) acc[t][r] *= corr[r];

    // ---- P: D-frag layout -> LDS -> A-frag layout (same-wave DS is in-order) ----
    unsigned short* pw = lds_p[wave];
#pragma unroll
    for (int r = 0; r < 8; ++r) {
      const int m = r + hi * 8;
      pw[m * VTS + ln]      = f2bf(s0[r]);
      pw[m * VTS + 16 + ln] = f2bf(s1[r]);
    }
    const unsigned short* pr = lds_p[wave] + ln * VTS;
    const v16bf ap = load_frag(pr + hi * 16, pr + 8 + hi * 16);

    // ---- row sums via P * ones (replaces 32 ds_bpermute reductions) ----
    v8f sums = wmma_bf16(ap, bones, z);
#pragma unroll
    for (int r = 0; r < 8; ++r) rsum[r] = rsum[r] * corr[r] + sums[r];

    // ---- context += P * V  (B-frags from transposed LDS tile) ----
#pragma unroll
    for (int t = 0; t < 4; ++t) {
      const unsigned short* vr = lds_vt + (t * 16 + ln) * VTS + hi * 16;
      acc[t] = wmma_bf16(ap, load_frag(vr, vr + 8), acc[t]);
    }
    __syncthreads();   // protect lds_vt / lds_k before next stage
  }

  // ---- normalize and store: ctx[(b*2048 + r)*1024 + h*64 + c] ----
  float inv[8];
#pragma unroll
  for (int r = 0; r < 8; ++r) inv[r] = 1.0f / rsum[r];
#pragma unroll
  for (int t = 0; t < 4; ++t) {
    const int c = h * HD + t * 16 + ln;
#pragma unroll
    for (int r = 0; r < 8; ++r) {
      const int m = m0 + r + hi * 8;
      ctx[(size_t)(b * SEQ + m) * EMB + c] = f2bf(acc[t][r] * inv[r]);
    }
  }
}

// ---------- launcher ----------

extern "C" void kernel_launch(void* const* d_in, const int* in_sizes, int n_in,
                              void* d_out, int out_size, void* d_ws, size_t ws_size,
                              hipStream_t stream) {
  const float* x     = (const float*)d_in[0];   // [2,2048,1024]
  const float* W_qkv = (const float*)d_in[1];   // [3072,1024]
  const float* b_qkv = (const float*)d_in[2];   // [3072]
  const float* W_out = (const float*)d_in[3];   // [1024,1024]
  const float* b_out = (const float*)d_in[4];   // [1024]
  float* out = (float*)d_out;                   // [2,2048,1024] f32

  const int M = BATCH * SEQ;                    // 4096
  unsigned short* Xb   = (unsigned short*)d_ws;                      // 4096*1024
  unsigned short* Wqb  = Xb  + (size_t)M * EMB;                      // 3072*1024
  unsigned short* Wob  = Wqb + (size_t)QKV3 * EMB;                   // 1024*1024
  unsigned short* QKVb = Wob + (size_t)EMB * EMB;                    // 4096*3072
  unsigned short* CTXb = QKVb + (size_t)M * QKV3;                    // 4096*1024

  // fp32 -> bf16 conversions
  {
    int np = (M * EMB) / 2;
    f32_to_bf16_kernel<<<(np + 255) / 256, 256, 0, stream>>>(x, (unsigned int*)Xb, np);
    np = (QKV3 * EMB) / 2;
    f32_to_bf16_kernel<<<(np + 255) / 256, 256, 0, stream>>>(W_qkv, (unsigned int*)Wqb, np);
    np = (EMB * EMB) / 2;
    f32_to_bf16_kernel<<<(np + 255) / 256, 256, 0, stream>>>(W_out, (unsigned int*)Wob, np);
  }

  // qkv = x @ W_qkv^T + b_qkv   (bf16 out)
  gemm_wmma_kernel<true><<<dim3(QKV3 / 64, M / 128), 256, 0, stream>>>(
      Xb, Wqb, b_qkv, (void*)QKVb, M, QKV3, EMB);

  // flash attention over the reshaped qkv -> ctx (bf16, already in output layout)
  attn_flash_kernel<<<dim3(SEQ / 64, HEADS, BATCH), 128, 0, stream>>>(QKVb, CTXb);

  // out = ctx @ W_out^T + b_out (f32 out)
  gemm_wmma_kernel<false><<<dim3(EMB / 64, M / 128), 256, 0, stream>>>(
      CTXb, Wob, b_out, (void*)out, M, EMB, EMB);
}
